// DynamicConv2d_6682969112830
// MI455X (gfx1250) — compile-verified, matrix operand order checked by reading, not run on hardware
//
#include <hip/hip_runtime.h>

// ---------------------------------------------------------------------------
// DynamicConv2d for MI455X (gfx1250, wave32, WMMA)
//
// Stage 1: per-sample global-avg-pool + 64x64 MLP -> hdn[32][64]      (f32)
// Stage 2: filter generation filt = hdn @ w2^T + b2, stored bf16 in
//          layout [b][tap j][cout][cin] (A-row-major per (b,j) GEMM tile)
// Stage 3: grouped conv as implicit GEMM per sample:
//          out[b] = filt_b[128 x 576] * im2col(x_b)[576 x 4096]
//          via v_wmma_f32_16x16x32_bf16. x is staged ONCE per block as a
//          halo'd LDS slab [4 rows][66 px][72 ch-slots]; im2col shifts are
//          pure LDS addressing. Filter tiles are double-buffered in LDS.
//          K-chunk loop fully unrolled -> constant-folded DS/global offsets.
//
// Sizes: B=32, CIN=64, COUT=128, K=3, H=W=64.
// ---------------------------------------------------------------------------

typedef __attribute__((ext_vector_type(16))) __bf16 bf16x16;
typedef __attribute__((ext_vector_type(8)))  __bf16 bf16x8;
typedef __attribute__((ext_vector_type(8)))  float  f32x8;

#define BATCH 32
#define CIN   64
#define COUT  128
#define HW    64
#define FILTD 64
#define NPER  73728            // COUT*CIN*9 outputs per sample
#define LDSTRIDE 40            // A tile: bf16/row (80B = 20 banks, 16B aligned)
#define XPX   66               // staged px per row incl. 1-px zero halo each side
#define XSTRIDE 72             // ch slots per px (144B = 36 dwords: conflict-free, 16B aligned)

// round-to-nearest-even f32 -> bf16 (bit pattern as ushort)
__device__ __forceinline__ unsigned short f2bf(float f) {
    unsigned int u = __float_as_uint(f);
    u += 0x7fffu + ((u >> 16) & 1u);
    return (unsigned short)(u >> 16);
}

// ---------------------------------------------------------------------------
// Kernel 1: global average pool + MLP layer 1 (relu). One block per sample.
// ---------------------------------------------------------------------------
__global__ __launch_bounds__(256) void pool_mlp1_kernel(
    const float* __restrict__ x, const float* __restrict__ w1,
    const float* __restrict__ b1, float* __restrict__ hdn)
{
    __shared__ float red[256];
    __shared__ float pooled[CIN];
    const int b = blockIdx.x;
    const int t = threadIdx.x;
    const int c = t >> 2;          // channel (4 threads per channel)
    const int q = t & 3;

    const float4* xb4 = (const float4*)(x + ((size_t)b * CIN + c) * (HW * HW));
    float s = 0.f;
    for (int i = q; i < 1024; i += 4) {          // 4096 floats as 1024 float4
        float4 v = xb4[i];
        s += (v.x + v.y) + (v.z + v.w);
    }
    red[t] = s;
    __syncthreads();
    if (t < CIN) {
        float p = (red[4 * t] + red[4 * t + 1] + red[4 * t + 2] + red[4 * t + 3])
                  * (1.0f / 4096.0f);
        pooled[t] = p;
    }
    __syncthreads();
    if (t < FILTD) {
        float a = b1[t];
        const float* w1r = w1 + t * CIN;
        #pragma unroll
        for (int k = 0; k < CIN; ++k) a += pooled[k] * w1r[k];
        hdn[b * FILTD + t] = fmaxf(a, 0.f);
    }
}

// ---------------------------------------------------------------------------
// Kernel 2: filter generation, bandwidth-bound on w2 (18.9 MB, read ONCE).
// One thread per output element o=(j*128+cout)*64+cin; 32 per-batch f32
// accumulators in VGPRs; hdn indices are wave-uniform -> scalar broadcasts.
// filt layout: [b][j][cout][cin] == b*73728 + o  (bf16)
// ---------------------------------------------------------------------------
__global__ __launch_bounds__(256) void gen_filters_kernel(
    const float* __restrict__ hdn, const float* __restrict__ w2,
    const float* __restrict__ b2, unsigned short* __restrict__ filt)
{
    const int o    = blockIdx.x * 256 + threadIdx.x;   // < 73728
    const int cin  = o & 63;
    const int cout = (o >> 6) & 127;
    const int j    = o >> 13;
    const int n    = cout * 576 + cin * 9 + j;         // row of w2 / index into b2

    const float* w2r = w2 + (size_t)n * FILTD;
    const float bias = b2[n];

    float acc[BATCH];
    #pragma unroll
    for (int b = 0; b < BATCH; ++b) acc[b] = bias;

    for (int k = 0; k < FILTD; k += 4) {
        float4 wv = *(const float4*)(w2r + k);
        #pragma unroll
        for (int kk = 0; kk < 4; ++kk) {
            const float w = (kk == 0) ? wv.x : (kk == 1) ? wv.y : (kk == 2) ? wv.z : wv.w;
            #pragma unroll
            for (int b = 0; b < BATCH; ++b)
                acc[b] += w * hdn[b * FILTD + k + kk];   // uniform -> SGPR broadcast
        }
    }
    #pragma unroll
    for (int b = 0; b < BATCH; ++b)
        filt[(size_t)b * NPER + o] = f2bf(acc[b]);
}

// ---------------------------------------------------------------------------
// Fragment load for v_wmma_f32_16x16x32_bf16 (ISA 7.12.2, wave32):
//   lane L<16 : row = L,    K = {0..7, 16..23}
//   lane L>=16: row = L-16, K = {8..15, 24..31}
// Caller passes the per-lane pointer (row base + kbase); both 8-element
// halves are 16B-aligned -> ds_load_b128 each.
// ---------------------------------------------------------------------------
__device__ __forceinline__ bf16x16 frag_ld(const unsigned short* p)
{
    const bf16x8* q = (const bf16x8*)p;
    bf16x8 lo = q[0];                                   // K = kb .. kb+7
    bf16x8 hi = q[2];                                   // K = kb+16 .. kb+23
    bf16x16 f;
    #pragma unroll
    for (int i = 0; i < 8; ++i) { f[i] = lo[i]; f[i + 8] = hi[i]; }
    return f;
}

// ---------------------------------------------------------------------------
// Kernel 3: implicit-GEMM grouped conv with WMMA.
// Block = (sample b, pixel-tile pt): M=128 couts x N=128 pixels (2 rows).
// 8 waves as 4(M) x 2(N); each wave: 2x4 16x16 f32 accum tiles.
// x slab staged once; A (filter) tiles double-buffered; 18 K-chunks unrolled.
// ---------------------------------------------------------------------------
__global__ __launch_bounds__(256) void dynconv_wmma_kernel(
    const float* __restrict__ x, const unsigned short* __restrict__ filt,
    float* __restrict__ out)
{
    __shared__ unsigned short Xs[4 * XPX * XSTRIDE];        // 38016 B, ch-innermost
    __shared__ unsigned short As[2][128 * LDSTRIDE];        // 2 x 10240 B

    const int b    = blockIdx.y;
    const int pt   = blockIdx.x;          // output rows 2*pt, 2*pt+1
    const int t    = threadIdx.x;
    const int lane = t & 31;
    const int wid  = t >> 5;
    const int wm   = wid & 3;             // M quadrant (32 couts)
    const int wn   = wid >> 2;            // N half (one output row of 64 px)

    f32x8 acc[2][4];
    #pragma unroll
    for (int i = 0; i < 2; ++i)
        #pragma unroll
        for (int n = 0; n < 4; ++n)
            acc[i][n] = f32x8{0.f, 0.f, 0.f, 0.f, 0.f, 0.f, 0.f, 0.f};

    // ---- stage x slab ONCE: rows 2pt-1 .. 2pt+2, all 64 channels, zero halo.
    {
        const int c = t & 63;             // channel, contiguous LDS writes
        const int r = t >> 6;             // staged row 0..3
        const int py = pt * 2 + r - 1;
        unsigned short* base = Xs + (size_t)(r * XPX) * XSTRIDE + c;
        if (py >= 0 && py < HW) {
            const float4* xr = (const float4*)(x + (((size_t)b * CIN + c) * HW + py) * HW);
            #pragma unroll
            for (int i4 = 0; i4 < 16; ++i4) {
                float4 v = xr[i4];
                const int px = 1 + i4 * 4;          // +1: halo offset
                base[(px + 0) * XSTRIDE] = f2bf(v.x);
                base[(px + 1) * XSTRIDE] = f2bf(v.y);
                base[(px + 2) * XSTRIDE] = f2bf(v.z);
                base[(px + 3) * XSTRIDE] = f2bf(v.w);
            }
        } else {
            #pragma unroll
            for (int px = 1; px <= 64; ++px) base[px * XSTRIDE] = 0;
        }
        base[0] = 0;                       // left halo
        base[65 * XSTRIDE] = 0;            // right halo
    }

    // A-tile loader role: 32B/thread fully-coalesced from filt[b][j][m][kc*32..]
    const int mA    = t >> 1;
    const int halfA = t & 1;
    auto stage_A = [&](int buf, int j, int kc) {
        const uint4* g = (const uint4*)(filt +
            ((((size_t)b * 9 + j) * COUT + mA) * CIN + kc * 32 + halfA * 16));
        uint4 v0 = g[0], v1 = g[1];
        uint4* l = (uint4*)(&As[buf][mA * LDSTRIDE + halfA * 16]);
        l[0] = v0; l[1] = v1;
    };

    const int kb  = (lane & 16) >> 1;      // 0 or 8 (fragment K base)
    const int l15 = lane & 15;

    stage_A(0, 0, 0);
    __syncthreads();                       // covers Xs staging + first A tile

    // fully unrolled: all tap/chunk indices and LDS offsets constant-fold
    #pragma unroll
    for (int chunk = 0; chunk < 18; ++chunk) {
        const int buf = chunk & 1;         // constant per unrolled iteration
        const int j  = chunk >> 1;
        const int kc = chunk & 1;
        const int dy = j / 3, dx = j % 3;

        // prefetch next A tile into the other buffer (overlaps the WMMAs)
        if (chunk + 1 < 18)
            stage_A(buf ^ 1, (chunk + 1) >> 1, (chunk + 1) & 1);

        // ---- compute: 8 WMMAs per wave per chunk (uniform flow, EXEC all-ones)
        const unsigned short* Ab = &As[buf][0];
        bf16x16 a0 = frag_ld(Ab + (wm * 32 +      l15) * LDSTRIDE + kb);
        bf16x16 a1 = frag_ld(Ab + (wm * 32 + 16 + l15) * LDSTRIDE + kb);

        // B fragments straight from the x slab: im2col shift = LDS addressing
        const unsigned short* Xrow =
            Xs + (size_t)((wn + dy) * XPX + dx + l15) * XSTRIDE + kc * 32 + kb;
        #pragma unroll
        for (int nt = 0; nt < 4; ++nt) {
            bf16x16 bf = frag_ld(Xrow + nt * 16 * XSTRIDE);
            acc[0][nt] = __builtin_amdgcn_wmma_f32_16x16x32_bf16(
                false, a0, false, bf, (short)0, acc[0][nt], false, false);
            acc[1][nt] = __builtin_amdgcn_wmma_f32_16x16x32_bf16(
                false, a1, false, bf, (short)0, acc[1][nt], false, false);
        }
        __syncthreads();   // one barrier/chunk: protects both buffers
    }

    // ---- epilogue: C/D layout: VGPR r, lane<16 -> M=r, lane>=16 -> M=r+8,
    //      N = lane&15. Lanes 0-15 write 64B contiguous per store.
    const int mofs = (lane & 16) >> 1;     // +8 for upper half-lanes
    const int py   = pt * 2 + wn;
    #pragma unroll
    for (int mt = 0; mt < 2; ++mt) {
        #pragma unroll
        for (int nt = 0; nt < 4; ++nt) {
            const int px = nt * 16 + l15;
            #pragma unroll
            for (int r = 0; r < 8; ++r) {
                const int m = wm * 32 + mt * 16 + mofs + r;
                out[(((size_t)b * COUT + m) * HW + py) * HW + px] = acc[mt][nt][r];
            }
        }
    }
}

// ---------------------------------------------------------------------------
extern "C" void kernel_launch(void* const* d_in, const int* in_sizes, int n_in,
                              void* d_out, int out_size, void* d_ws, size_t ws_size,
                              hipStream_t stream)
{
    const float* x  = (const float*)d_in[0];   // [32,64,64,64]
    const float* w1 = (const float*)d_in[1];   // [64,64]
    const float* b1 = (const float*)d_in[2];   // [64]
    const float* w2 = (const float*)d_in[3];   // [73728,64]
    const float* b2 = (const float*)d_in[4];   // [73728]
    float* out = (float*)d_out;                // [32,128,64,64]

    // workspace: hdn f32 (8 KB) then bf16 filters [b][j][cout][cin] (4.5 MB)
    float* hdn = (float*)d_ws;
    unsigned short* filt = (unsigned short*)((char*)d_ws + 8192);

    pool_mlp1_kernel<<<BATCH, 256, 0, stream>>>(x, w1, b1, hdn);
    gen_filters_kernel<<<NPER / 256, 256, 0, stream>>>(hdn, w2, b2, filt);
    dynconv_wmma_kernel<<<dim3(HW / 2, BATCH), 256, 0, stream>>>(x, filt, out);
}
